// FixedRevRNN_3478923510439
// MI455X (gfx1250) — matrix operand order repaired and checked by reading, not compiled, not run on hardware
//
#include <hip/hip_runtime.h>
#include <hip/hip_bf16.h>

// ---------------------------------------------------------------------------
// FixedRevRNN for MI455X (gfx1250, wave32, WMMA).
//
//   * Hoist recurrence-independent GEMMs  s@p0b / s@p1b  into two large
//     parallel bf16 WMMA GEMMs (M = T*B = 16384), gathered straight from the
//     embedding table.
//   * All 1024x1024 weights converted once to bf16 in the WMMA B-operand
//     (N-major) layout: 12 MB total -> L2-resident; the 1024 sequential
//     per-step GEMMs stream weights from L2, not HBM.
//   * Sequential core per step: act_norm -> WMMA GEMM (+precomputed s@B)
//     -> act_norm -> WMMA GEMM -> fused QR(32x32 MGS via wave shuffles) +
//     h@Q state update + bf16 feature writeout.
//   * GEMM k-loops are software-pipelined (double-buffered fragments,
//     unroll-by-2) so global loads overlap WMMA issue instead of draining
//     loadcnt to zero before every matrix op.
//   * Final projection: [16384,2048]x[2048,256] bf16 WMMA GEMM with bias,
//     nontemporal store (logits are write-once).
// ---------------------------------------------------------------------------

typedef __bf16 bf16_t;
typedef __attribute__((ext_vector_type(16))) __bf16 v16bf;
typedef __attribute__((ext_vector_type(8)))  float  v8f;

union AFrag { v16bf v; unsigned int u[8]; };
union BFrag { v16bf v; uint4 q[2]; };

__device__ __forceinline__ unsigned short f2bf(float f) {
  unsigned int u = __float_as_uint(f);
  u += 0x7FFFu + ((u >> 16) & 1u);           // round-to-nearest-even
  return (unsigned short)(u >> 16);
}
__device__ __forceinline__ float bf2f(unsigned short h) {
  return __uint_as_float(((unsigned int)h) << 16);
}
__device__ __forceinline__ float waveReduce(float v) {
  #pragma unroll
  for (int off = 16; off > 0; off >>= 1) v += __shfl_xor(v, off, 32);
  return v;
}

// A-fragment (16x32 bf16): packed K-pairs per dword, contiguous K;
// lanes 16..31 hold K+16..K+31.  p = row base + k0 dwords.
__device__ __forceinline__ void load_a(AFrag& a, const unsigned int* __restrict__ p,
                                       int kh) {
  #pragma unroll
  for (int v = 0; v < 8; ++v) {
    const int kb = (v << 1) + kh;
    a.u[v] = p[kb >> 1];
  }
}

// ---------------------------------------------------------------------------
// Generic bf16 WMMA GEMM:  C[M,N] (+= Cadd) = A[M,K] * Bt[N,K]^T
//   Each wave computes a 16x64 slab (4 accumulators, A-frag reused 4x);
//   k-loop double-buffered.
// ---------------------------------------------------------------------------
template<bool GATHER, bool ADDC, bool OUTBF>
__global__ void __launch_bounds__(256)
gemm_bf16_wmma(const unsigned short* __restrict__ A,
               const int*            __restrict__ gidx,   // [64][256] when GATHER
               const unsigned short* __restrict__ Bt,     // [N][K] bf16 (N-major)
               const unsigned short* __restrict__ Cadd,   // [M][N] bf16 when ADDC
               float*                __restrict__ Cf,
               unsigned short*       __restrict__ Cbf,
               int M, int N, int K)
{
  const int lane = threadIdx.x & 31;
  const int nT4  = N >> 6;                                // groups of 4 N-tiles
  const int g    = blockIdx.x * 8 + (threadIdx.x >> 5);
  if (g >= (M >> 4) * nT4) return;                        // wave-uniform exit
  const int row0 = (g / nT4) << 4;
  const int col0 = (g % nT4) << 6;

  const int mA = row0 + (lane & 15);
  int arow;
  if (GATHER) arow = gidx[(mA & 63) * 256 + (mA >> 6)];   // m = t*64 + b
  else        arow = mA;
  const unsigned int* __restrict__ ap =
      (const unsigned int*)(A + (size_t)arow * K);
  const int kh = (lane >> 4) << 4;                        // A K-half select
  const size_t bstep = (size_t)(K << 1);                  // 16 rows of Bt in uint4
  const unsigned short* __restrict__ bbase =
      Bt + (size_t)(col0 + (lane & 15)) * K + ((lane >> 4) << 4);

  auto loadB4 = [&](BFrag* b, int k0) {
    const uint4* __restrict__ qb = (const uint4*)(bbase + k0);
    #pragma unroll
    for (int j = 0; j < 4; ++j) {
      b[j].q[0] = qb[j * bstep];
      b[j].q[1] = qb[j * bstep + 1];
    }
  };

  v8f acc[4] = {};
  AFrag a0, a1;
  BFrag b0[4], b1[4];
  load_a(a0, ap, kh);
  loadB4(b0, 0);
  for (int k0 = 0; k0 < K; k0 += 64) {                    // K % 64 == 0
    load_a(a1, ap + ((k0 + 32) >> 1), kh);                // prefetch buf 1
    loadB4(b1, k0 + 32);
    #pragma unroll
    for (int j = 0; j < 4; ++j)
      acc[j] = __builtin_amdgcn_wmma_f32_16x16x32_bf16(
          false, a0.v, false, b0[j].v, (short)0, acc[j], false, false);
    if (k0 + 64 < K) {                                    // prefetch buf 0
      load_a(a0, ap + ((k0 + 64) >> 1), kh);
      loadB4(b0, k0 + 64);
    }
    #pragma unroll
    for (int j = 0; j < 4; ++j)
      acc[j] = __builtin_amdgcn_wmma_f32_16x16x32_bf16(
          false, a1.v, false, b1[j].v, (short)0, acc[j], false, false);
  }

  const int mb = row0 + ((lane >> 4) << 3);
  const int nb = lane & 15;
  #pragma unroll
  for (int j = 0; j < 4; ++j) {
    const int n = col0 + (j << 4) + nb;
    #pragma unroll
    for (int r = 0; r < 8; ++r) {
      float val = acc[j][r];
      const int mm = mb + r;
      if (ADDC)  val += bf2f(Cadd[(size_t)mm * N + n]);
      if (OUTBF) Cbf[(size_t)mm * N + n] = f2bf(val);
      else       Cf [(size_t)mm * N + n] = val;
    }
  }
}

// ---------------------------------------------------------------------------
// Output projection: logits[b,o,t] = sum_c feats[b,t,c]*W[o,c] + bias[o]
//   A row m = b*256 + t ;  c<1024 -> YS0[t,b,:], else YS1[t,b,:]
//   out_w is already [O=256][C=2048] row-major == the N-major B layout.
// ---------------------------------------------------------------------------
__global__ void __launch_bounds__(256)
proj_gemm_wmma(const unsigned short* __restrict__ YS0,
               const unsigned short* __restrict__ YS1,
               const unsigned short* __restrict__ Wb,     // [256][2048] bf16
               const float*          __restrict__ bias,   // [256]
               float*                __restrict__ out)    // [64][256][256]
{
  const int lane = threadIdx.x & 31;
  const int g    = blockIdx.x * 8 + (threadIdx.x >> 5);   // 4096 waves total
  const int row0 = (g >> 2) << 4;                         // M = 16384
  const int col0 = (g & 3) << 6;                          // N = 256

  const int mA = row0 + (lane & 15);
  const size_t arow = ((size_t)(mA & 255) * 64 + (mA >> 8)) * 1024;
  const unsigned int* __restrict__ a0p = (const unsigned int*)(YS0 + arow);
  const unsigned int* __restrict__ a1p = (const unsigned int*)(YS1 + arow);
  const int kh = (lane >> 4) << 4;
  const size_t bstep = (size_t)(2048 << 1);
  const unsigned short* __restrict__ bbase =
      Wb + (size_t)(col0 + (lane & 15)) * 2048 + ((lane >> 4) << 4);

  auto apk = [&](int k) {
    return (k < 1024) ? (a0p + (k >> 1)) : (a1p + ((k - 1024) >> 1));
  };
  auto loadB4 = [&](BFrag* b, int k0) {
    const uint4* __restrict__ qb = (const uint4*)(bbase + k0);
    #pragma unroll
    for (int j = 0; j < 4; ++j) {
      b[j].q[0] = qb[j * bstep];
      b[j].q[1] = qb[j * bstep + 1];
    }
  };

  v8f acc[4] = {};
  AFrag a0, a1;
  BFrag b0[4], b1[4];
  load_a(a0, apk(0), kh);
  loadB4(b0, 0);
  for (int k0 = 0; k0 < 2048; k0 += 64) {
    load_a(a1, apk(k0 + 32), kh);
    loadB4(b1, k0 + 32);
    #pragma unroll
    for (int j = 0; j < 4; ++j)
      acc[j] = __builtin_amdgcn_wmma_f32_16x16x32_bf16(
          false, a0.v, false, b0[j].v, (short)0, acc[j], false, false);
    if (k0 + 64 < 2048) {
      load_a(a0, apk(k0 + 64), kh);
      loadB4(b0, k0 + 64);
    }
    #pragma unroll
    for (int j = 0; j < 4; ++j)
      acc[j] = __builtin_amdgcn_wmma_f32_16x16x32_bf16(
          false, a1.v, false, b1[j].v, (short)0, acc[j], false, false);
  }

  const int mb = row0 + ((lane >> 4) << 3);
  const int nb = lane & 15;
  #pragma unroll
  for (int j = 0; j < 4; ++j) {
    const int n = col0 + (j << 4) + nb;
    const float bs = bias[n];
    #pragma unroll
    for (int r = 0; r < 8; ++r) {
      const int mm = mb + r;                              // m = b*256 + t
      __builtin_nontemporal_store(
          acc[j][r] + bs,
          &out[((size_t)(mm >> 8) * 256 + n) * 256 + (mm & 255)]);
    }
  }
}

// ---------------------------------------------------------------------------
// activate_norm: relu -> mean-center -> *32/(||y||+eps), f32 in, bf16 out.
// One block (256 threads) per row of 1024.
// ---------------------------------------------------------------------------
__global__ void __launch_bounds__(256)
act_norm_kernel(const float* __restrict__ X, unsigned short* __restrict__ Y)
{
  const int row = blockIdx.x, tid = threadIdx.x;
  const int lane = tid & 31, wid = tid >> 5;
  const float* x = X + (size_t)row * 1024;
  __shared__ float sred[8];

  float v[4];
  float s = 0.f;
  #pragma unroll
  for (int i = 0; i < 4; ++i) {
    float t = x[tid + 256 * i];
    v[i] = t > 0.f ? t : 0.f;
    s += v[i];
  }
  float ws = waveReduce(s);
  if (lane == 0) sred[wid] = ws;
  __syncthreads();
  float tot = 0.f;
  #pragma unroll
  for (int i = 0; i < 8; ++i) tot += sred[i];
  const float mean = tot * (1.0f / 1024.0f);
  __syncthreads();

  float ss = 0.f;
  #pragma unroll
  for (int i = 0; i < 4; ++i) { v[i] -= mean; ss += v[i] * v[i]; }
  float ws2 = waveReduce(ss);
  if (lane == 0) sred[wid] = ws2;
  __syncthreads();
  float tss = 0.f;
  #pragma unroll
  for (int i = 0; i < 8; ++i) tss += sred[i];
  const float scale = 32.0f / (sqrtf(tss) + 1e-5f);
  #pragma unroll
  for (int i = 0; i < 4; ++i)
    Y[(size_t)row * 1024 + tid + 256 * i] = f2bf(v[i] * scale);
}

// ---------------------------------------------------------------------------
// Fused QR (32x32 modified Gram-Schmidt) + state update H[b] = H[b] @ Q,
// + bf16 feature writeout. One wave per batch element; lane = matrix row.
// Cross-lane dots via __shfl_xor; Q shared through LDS for the h@Q product.
// ---------------------------------------------------------------------------
__global__ void __launch_bounds__(256)
qr_update_kernel(const float* __restrict__ O,     // [64][1024] pre-QR
                 float*       __restrict__ H,     // [64][1024] state (in/out)
                 unsigned short* __restrict__ YS) // [64][1024] bf16 feats out
{
  __shared__ float sh[256][33];
  const int tid = threadIdx.x, lane = tid & 31, w = tid >> 5;
  const int b = blockIdx.x * 8 + w;

  const float* ob = O + (size_t)b * 1024 + lane * 32;
  #pragma unroll
  for (int k = 0; k < 32; ++k) sh[tid][k] = ob[k];

  for (int j = 0; j < 32; ++j) {                  // modified Gram-Schmidt
    float vj  = sh[tid][j];
    float n2  = waveReduce(vj * vj);
    float inv = (n2 > 1e-24f) ? rsqrtf(n2) : 0.0f;
    float qj  = vj * inv;
    sh[tid][j] = qj;
    for (int k = j + 1; k < 32; ++k) {
      float vk = sh[tid][k];
      float r  = waveReduce(qj * vk);
      sh[tid][k] = vk - r * qj;
    }
  }
  __syncthreads();

  float* hb = H + (size_t)b * 1024 + lane * 32;
  float hrow[32];
  #pragma unroll
  for (int k = 0; k < 32; ++k) hrow[k] = hb[k];

  unsigned short* ys = YS + (size_t)b * 1024 + lane * 32;
  #pragma unroll 4
  for (int j = 0; j < 32; ++j) {
    float acc = 0.f;
    #pragma unroll
    for (int k = 0; k < 32; ++k)
      acc = fmaf(hrow[k], sh[(w << 5) + k][j], acc);
    hb[j] = acc;
    ys[j] = f2bf(acc);
  }
}

// ---------------------------------------------------------------------------
// Setup kernels.
// ---------------------------------------------------------------------------
__global__ void __launch_bounds__(256)
transpose_to_bf16(const float* __restrict__ W, unsigned short* __restrict__ Wt)
{
  __shared__ float tile[32][33];
  const int tx = threadIdx.x & 31, ty = threadIdx.x >> 5;
  const int gx = blockIdx.x * 32, gy = blockIdx.y * 32;
  #pragma unroll
  for (int i = 0; i < 4; ++i)
    tile[ty + 8 * i][tx] = W[(size_t)(gy + ty + 8 * i) * 1024 + gx + tx];
  __syncthreads();
  #pragma unroll
  for (int i = 0; i < 4; ++i)
    Wt[(size_t)(gx + ty + 8 * i) * 1024 + gy + tx] = f2bf(tile[tx][ty + 8 * i]);
}

__global__ void __launch_bounds__(256)
conv_bf16(const float* __restrict__ X, unsigned short* __restrict__ Y, int n)
{
  const int i = blockIdx.x * 256 + threadIdx.x;
  if (i < n) Y[i] = f2bf(X[i]);
}

__global__ void __launch_bounds__(256)
init_state(const float* __restrict__ h, float* __restrict__ H)
{
  const int i = blockIdx.x * 256 + threadIdx.x;   // 64*1024 elements
  H[i] = h[i & 1023];
}

// ---------------------------------------------------------------------------
// Host launcher.
// ---------------------------------------------------------------------------
extern "C" void kernel_launch(void* const* d_in, const int* in_sizes, int n_in,
                              void* d_out, int out_size, void* d_ws, size_t ws_size,
                              hipStream_t stream)
{
  (void)in_sizes; (void)n_in; (void)out_size; (void)ws_size;
  const int*   fn   = (const int*)  d_in[0];   // [64][256]
  const float* wsrc0= (const float*)d_in[1];   // p0a
  const float* wsrc1= (const float*)d_in[2];   // p0b
  const float* wsrc2= (const float*)d_in[3];   // p0c
  const float* wsrc3= (const float*)d_in[4];   // p1a
  const float* wsrc4= (const float*)d_in[5];   // p1b
  const float* wsrc5= (const float*)d_in[6];   // p1c
  const float* emb  = (const float*)d_in[7];   // [256][1024]
  const float* outw = (const float*)d_in[8];   // [256][2048]
  const float* outb = (const float*)d_in[9];   // [256]
  const float* h0   = (const float*)d_in[10];  // [32][32]
  const float* h1   = (const float*)d_in[11];  // [32][32]

  char* wsp = (char*)d_ws;
  auto alloc = [&](size_t bytes) -> char* {
    char* p = wsp; wsp += (bytes + 255) & ~(size_t)255; return p;
  };
  unsigned short* Wt[6];
  for (int i = 0; i < 6; ++i) Wt[i] = (unsigned short*)alloc((size_t)1024 * 1024 * 2);
  unsigned short* WoutBf = (unsigned short*)alloc((size_t)256 * 2048 * 2);
  unsigned short* EmbBf  = (unsigned short*)alloc((size_t)256 * 1024 * 2);
  unsigned short* SB0    = (unsigned short*)alloc((size_t)16384 * 1024 * 2);
  unsigned short* SB1    = (unsigned short*)alloc((size_t)16384 * 1024 * 2);
  unsigned short* YS0    = (unsigned short*)alloc((size_t)16384 * 1024 * 2);
  unsigned short* YS1    = (unsigned short*)alloc((size_t)16384 * 1024 * 2);
  float* H0   = (float*)alloc((size_t)64 * 1024 * 4);
  float* H1   = (float*)alloc((size_t)64 * 1024 * 4);
  float* T0   = (float*)alloc((size_t)64 * 1024 * 4);
  float* Obuf = (float*)alloc((size_t)64 * 1024 * 4);
  unsigned short* XN = (unsigned short*)alloc((size_t)64 * 1024 * 2);

  // --- setup: weight conversion (bf16, WMMA-B layout), states, embeddings ---
  const float* Wsrc[6] = { wsrc0, wsrc1, wsrc2, wsrc3, wsrc4, wsrc5 };
  for (int i = 0; i < 6; ++i)
    transpose_to_bf16<<<dim3(32, 32), dim3(256), 0, stream>>>(Wsrc[i], Wt[i]);
  conv_bf16<<<(256 * 2048) / 256, 256, 0, stream>>>(outw, WoutBf, 256 * 2048);
  conv_bf16<<<(256 * 1024) / 256, 256, 0, stream>>>(emb,  EmbBf,  256 * 1024);
  init_state<<<256, 256, 0, stream>>>(h0, H0);
  init_state<<<256, 256, 0, stream>>>(h1, H1);

  // --- hoisted input GEMMs: SBx[t*64+b, :] = embed[fn[b,t]] @ pXb ---
  gemm_bf16_wmma<true, false, true><<<2048, 256, 0, stream>>>(
      EmbBf, fn, Wt[1], nullptr, nullptr, SB0, 16384, 1024, 1024);
  gemm_bf16_wmma<true, false, true><<<2048, 256, 0, stream>>>(
      EmbBf, fn, Wt[4], nullptr, nullptr, SB1, 16384, 1024, 1024);

  // --- sequential scan: 256 steps, two reversible half-steps each ---
  for (int t = 0; t < 256; ++t) {
    const size_t off = (size_t)t * 64 * 1024;
    // phase A: Q0 = qr(calc(H0)); H1 = H1 @ Q0 ; ys1[t] = H1
    act_norm_kernel<<<64, 256, 0, stream>>>(H0, XN);
    gemm_bf16_wmma<false, true, false><<<8, 256, 0, stream>>>(
        XN, nullptr, Wt[0], SB0 + off, T0, nullptr, 64, 1024, 1024);
    act_norm_kernel<<<64, 256, 0, stream>>>(T0, XN);
    gemm_bf16_wmma<false, false, false><<<8, 256, 0, stream>>>(
        XN, nullptr, Wt[2], nullptr, Obuf, nullptr, 64, 1024, 1024);
    qr_update_kernel<<<8, 256, 0, stream>>>(Obuf, H1, YS1 + off);
    // phase B: Q1 = qr(calc(H1)); H0 = H0 @ Q1 ; ys0[t] = H0
    act_norm_kernel<<<64, 256, 0, stream>>>(H1, XN);
    gemm_bf16_wmma<false, true, false><<<8, 256, 0, stream>>>(
        XN, nullptr, Wt[3], SB1 + off, T0, nullptr, 64, 1024, 1024);
    act_norm_kernel<<<64, 256, 0, stream>>>(T0, XN);
    gemm_bf16_wmma<false, false, false><<<8, 256, 0, stream>>>(
        XN, nullptr, Wt[5], nullptr, Obuf, nullptr, 64, 1024, 1024);
    qr_update_kernel<<<8, 256, 0, stream>>>(Obuf, H0, YS0 + off);
  }

  // --- output projection with bias -> d_out [64][256][256] f32 ---
  proj_gemm_wmma<<<512, 256, 0, stream>>>(YS0, YS1, WoutBf, outb, (float*)d_out);
}
